// InterpretableMultiHeadAttention_23776938950717
// MI455X (gfx1250) — matrix operand order repaired and checked by reading, not compiled, not run on hardware
//
#include <hip/hip_runtime.h>

typedef __bf16 bf16;
typedef __attribute__((ext_vector_type(16))) __bf16 v16bf;
typedef __attribute__((ext_vector_type(8)))  __bf16 v8bf;
typedef __attribute__((ext_vector_type(8)))  float  v8f;
typedef unsigned int uint32x4 __attribute__((ext_vector_type(4)));
typedef int          int32x8  __attribute__((ext_vector_type(8)));
typedef int          int32x4  __attribute__((ext_vector_type(4)));

#define DMODEL 1024
#define NHEADS 16
#define DK     64
#define BATCH  2
#define SEQ    2048
#define NROWS  (BATCH * SEQ)   // 4096

// Padded LDS row stride for staged A tile: 1024 elems + 2 pads of 8 elems
// (TDM pad: 4 dwords after every 256 dwords) -> 1040 bf16 per row.
#define A_LDS_STRIDE 1040

// ---------- WMMA helpers ----------

static __device__ __forceinline__ v8f zero8() {
  v8f z = {0.f, 0.f, 0.f, 0.f, 0.f, 0.f, 0.f, 0.f};
  return z;
}

static __device__ __forceinline__ v8f wmma_bf16(v16bf a, v16bf b, v8f c) {
  return __builtin_amdgcn_wmma_f32_16x16x32_bf16(
      /*neg_a=*/false, a, /*neg_b=*/false, b,
      /*c_mod=*/(short)0, c, /*reuse_a=*/false, /*reuse_b=*/false);
}

// A-matrix 16x32 bf16 frag from global: lane r=lane&15 holds row r; half
// selects K chunks [8*half,+8) and [16+8*half,+8). (ISA 7.12.2)
static __device__ __forceinline__ v16bf fragA_g(const bf16* base, int ld) {
  const int lane = threadIdx.x & 31;
  const int r = lane & 15, h = (lane >> 4) & 1;
  const bf16* p = base + (size_t)r * ld + h * 8;
  v8bf lo = *(const v8bf*)(p);
  v8bf hi = *(const v8bf*)(p + 16);
  return __builtin_shufflevector(lo, hi, 0, 1, 2, 3, 4, 5, 6, 7,
                                 8, 9, 10, 11, 12, 13, 14, 15);
}

// Same A frag sourced from the TDM-staged, pad-interleaved LDS tile.
// Element (r,k) lives at r*A_LDS_STRIDE + k + (k>=512 ? 8 : 0).
static __device__ __forceinline__ v16bf fragA_lds(const bf16* sA, int r0, int k0) {
  const int lane = threadIdx.x & 31;
  const int r = r0 + (lane & 15), h = (lane >> 4) & 1;
  const int k1 = k0 + h * 8;
  const int k2 = k0 + 16 + h * 8;
  const bf16* p1 = sA + r * A_LDS_STRIDE + k1 + (k1 >= 512 ? 8 : 0);
  const bf16* p2 = sA + r * A_LDS_STRIDE + k2 + (k2 >= 512 ? 8 : 0);
  v8bf lo = *(const v8bf*)p1;
  v8bf hi = *(const v8bf*)p2;
  return __builtin_shufflevector(lo, hi, 0, 1, 2, 3, 4, 5, 6, 7,
                                 8, 9, 10, 11, 12, 13, 14, 15);
}

// B-matrix 32x16 bf16 frag: lane c=lane&15 holds column c; half selects
// K chunks [8*half,+8) and [16+8*half,+8). `ld` = stride between columns.
static __device__ __forceinline__ v16bf fragB_g(const bf16* base, int ld) {
  const int lane = threadIdx.x & 31;
  const int c = lane & 15, h = (lane >> 4) & 1;
  const bf16* p = base + (size_t)c * ld + h * 8;
  v8bf lo = *(const v8bf*)(p);
  v8bf hi = *(const v8bf*)(p + 16);
  return __builtin_shufflevector(lo, hi, 0, 1, 2, 3, 4, 5, 6, 7,
                                 8, 9, 10, 11, 12, 13, 14, 15);
}

// A-frag sourced from a 16x32 row-major f32 LDS tile, converting to bf16.
static __device__ __forceinline__ v16bf fragA_lds_f32(const float* buf) {
  const int lane = threadIdx.x & 31;
  const int r = lane & 15, h = (lane >> 4) & 1;
  const float* p = buf + r * 32 + h * 8;
  v16bf f;
#pragma unroll
  for (int i = 0; i < 8; ++i) {
    f[i]     = (bf16)p[i];
    f[i + 8] = (bf16)p[16 + i];
  }
  return f;
}

// The 4 B-frags covering one 16x32 score tile's K-matrix inputs.
struct KFrags { v16bf f[4]; };

static __device__ __forceinline__ void loadK(KFrags& kf, const bf16* Kh,
                                             int bh, int kv0) {
  const bf16* Kbase = Kh + (((size_t)bh) * SEQ + kv0) * DK;
  kf.f[0] = fragB_g(Kbase, DK);
  kf.f[1] = fragB_g(Kbase + 32, DK);
  kf.f[2] = fragB_g(Kbase + 16 * DK, DK);
  kf.f[3] = fragB_g(Kbase + 16 * DK + 32, DK);
}

// ---------- Kernels ----------

__global__ void cvt_kernel(const float* __restrict__ in, bf16* __restrict__ out, int n) {
  int i = blockIdx.x * blockDim.x + threadIdx.x;
  if (i < n) out[i] = (bf16)in[i];
}

// C = A(bf16 [4096x1024]) @ W(bf16 [1024x1024])^T.
// Block = 32 rows x 512 cols; wave computes a 32x64 tile (8 C-tiles).
// A row-tile is staged via the Tensor Data Mover into LDS once per block.
// mode 0: bf16 out [b,h,s,d]   (Q, K)
// mode 1: bf16 out [b,h,d,s]   (V transposed per head)
// mode 2: f32  out [n, 1024]   (final projection)
__global__ void __launch_bounds__(256)
gemm_kernel(const bf16* __restrict__ A, const bf16* __restrict__ W,
            void* __restrict__ outp, int mode) {
  __shared__ __align__(16) bf16 sA[32 * A_LDS_STRIDE];

  const int wave = threadIdx.x >> 5;
  const int lane = threadIdx.x & 31;
  const int half = (lane >> 4) & 1, ln = lane & 15;
  const int row0 = blockIdx.x * 32;
  const int col0 = blockIdx.y * 512 + wave * 64;

  // --- TDM: stage A[row0:row0+32, 0:1024] -> LDS (pad 4dw per 256dw) ---
  if (threadIdx.x < 32) {
    const unsigned int lds_addr = (unsigned int)(size_t)(&sA[0]);
    const unsigned long long gaddr =
        (unsigned long long)(size_t)(A + (size_t)row0 * DMODEL);
    uint32x4 g0;
    g0[0] = 1u;                                  // count=1 (valid user D#)
    g0[1] = lds_addr;                            // lds_addr (bytes)
    g0[2] = (unsigned int)gaddr;                 // global_addr lo32
    g0[3] = (unsigned int)((gaddr >> 32) & 0x1FFFFFFull) | (2u << 30); // +type=2
    int32x8 g1;
    g1[0] = (int)((1u << 16)        // data_size = 2 bytes
                | (1u << 20)        // pad_enable
                | (7u << 22)        // pad_interval: 256 dwords
                | (3u << 25));      // pad_amount: 4 dwords
    g1[1] = (int)((unsigned)(DMODEL & 0xFFFF) << 16);    // tensor_dim0 lo16
    g1[2] = (int)((DMODEL >> 16) | ((unsigned)(NROWS & 0xFFFF) << 16)); // dim0 hi | dim1 lo
    g1[3] = (int)((NROWS >> 16) | ((unsigned)DMODEL << 16));            // dim1 hi | tile_dim0
    g1[4] = 32;                                          // tile_dim1=32, tile_dim2=0
    g1[5] = DMODEL;                                      // tensor_dim0_stride lo32
    g1[6] = 0;                                           // stride hi | dim1_stride lo
    g1[7] = 0;
    int32x4 gz = {0, 0, 0, 0};
#if __clang_major__ >= 23
    int32x8 gz8 = {0, 0, 0, 0, 0, 0, 0, 0};
    __builtin_amdgcn_tensor_load_to_lds(g0, g1, gz, gz, gz8, 0);
#else
    __builtin_amdgcn_tensor_load_to_lds(g0, g1, gz, gz, 0);
#endif
    __builtin_amdgcn_s_wait_tensorcnt(0);
  }
  __syncthreads();

  v8f acc[8];
#pragma unroll
  for (int t = 0; t < 8; ++t) acc[t] = zero8();

  const bf16* Wc0 = W + (size_t)(col0 + 0) * DMODEL;
  const bf16* Wc1 = W + (size_t)(col0 + 16) * DMODEL;
  const bf16* Wc2 = W + (size_t)(col0 + 32) * DMODEL;
  const bf16* Wc3 = W + (size_t)(col0 + 48) * DMODEL;

  // Software pipeline: prefetch next chunk's frags into distinct registers.
  v16bf a0 = fragA_lds(sA, 0, 0);
  v16bf a1 = fragA_lds(sA, 16, 0);
  v16bf b0 = fragB_g(Wc0, DMODEL);
  v16bf b1 = fragB_g(Wc1, DMODEL);
  v16bf b2 = fragB_g(Wc2, DMODEL);
  v16bf b3 = fragB_g(Wc3, DMODEL);

  for (int kc = 0; kc < DMODEL / 32; ++kc) {
    v16bf na0, na1, nb0, nb1, nb2, nb3;
    if (kc < DMODEL / 32 - 1) {
      const int k0 = (kc + 1) * 32;
      nb0 = fragB_g(Wc0 + k0, DMODEL);
      nb1 = fragB_g(Wc1 + k0, DMODEL);
      nb2 = fragB_g(Wc2 + k0, DMODEL);
      nb3 = fragB_g(Wc3 + k0, DMODEL);
      na0 = fragA_lds(sA, 0, k0);
      na1 = fragA_lds(sA, 16, k0);
      if (kc < DMODEL / 32 - 2) {
        const int kp = (kc + 2) * 32;
        __builtin_prefetch(Wc0 + (size_t)ln * DMODEL + kp, 0, 3);
        __builtin_prefetch(Wc2 + (size_t)ln * DMODEL + kp, 0, 3);
      }
    }
    acc[0] = wmma_bf16(a0, b0, acc[0]);
    acc[1] = wmma_bf16(a0, b1, acc[1]);
    acc[2] = wmma_bf16(a0, b2, acc[2]);
    acc[3] = wmma_bf16(a0, b3, acc[3]);
    acc[4] = wmma_bf16(a1, b0, acc[4]);
    acc[5] = wmma_bf16(a1, b1, acc[5]);
    acc[6] = wmma_bf16(a1, b2, acc[6]);
    acc[7] = wmma_bf16(a1, b3, acc[7]);
    if (kc < DMODEL / 32 - 1) {
      a0 = na0; a1 = na1;
      b0 = nb0; b1 = nb1; b2 = nb2; b3 = nb3;
    }
  }

#pragma unroll
  for (int rh = 0; rh < 2; ++rh) {
#pragma unroll
    for (int nt = 0; nt < 4; ++nt) {
      const v8f c = acc[rh * 4 + nt];
#pragma unroll
      for (int i = 0; i < 8; ++i) {
        const int row = row0 + rh * 16 + i + half * 8;
        const int col = col0 + nt * 16 + ln;
        const float v = c[i];
        if (mode == 2) {
          ((float*)outp)[(size_t)row * DMODEL + col] = v;
        } else {
          const int bb = row >> 11, s = row & (SEQ - 1);
          const int h = col >> 6, d = col & (DK - 1);
          bf16* o = (bf16*)outp;
          if (mode == 0)
            o[(((size_t)(bb * NHEADS + h)) * SEQ + s) * DK + d] = (bf16)v;
          else
            o[(((size_t)(bb * NHEADS + h)) * DK + d) * SEQ + s] = (bf16)v;
        }
      }
    }
  }
}

// Two-pass flash attention. Block = (batch b, 16-query tile). 8 waves, each
// owns 2 heads. K-frags are software-pipelined at (j,head) granularity and
// V-frags are hoisted ahead of the softmax/LDS work so the WMMAs don't sit
// behind full-latency waits. Emits attended (bf16 [n,1024]) and
// avg_attention (f32 [B,S,S]).
__global__ void __launch_bounds__(256)
attn_kernel(const bf16* __restrict__ Qh, const bf16* __restrict__ Kh,
            const bf16* __restrict__ Vt, bf16* __restrict__ Att,
            float* __restrict__ Avg) {
  __shared__ float s_p[16][512];  // 16 head-buffers of 16x32 probs (32 KB)

  const int qt = blockIdx.x, b = blockIdx.y;
  const int q0 = qt * 16;
  const int wave = threadIdx.x >> 5;
  const int lane = threadIdx.x & 31;
  const int half = (lane >> 4) & 1, ln = lane & 15;
  const float scale = 0.125f;  // 1/sqrt(DK)
  const int bh0 = b * NHEADS + wave * 2;

  v16bf qa[2][2];
#pragma unroll
  for (int hs = 0; hs < 2; ++hs) {
    const bf16* Qbase = Qh + (((size_t)(bh0 + hs)) * SEQ + q0) * DK;
    qa[hs][0] = fragA_g(Qbase + 0, DK);
    qa[hs][1] = fragA_g(Qbase + 32, DK);
  }

  // ---- Pass 1: row max + sum-exp (rows i+8*half live in elem i of C frags)
  float m[2][8], l[2][8];
#pragma unroll
  for (int hs = 0; hs < 2; ++hs)
#pragma unroll
    for (int i = 0; i < 8; ++i) { m[hs][i] = -3.0e38f; l[hs][i] = 0.f; }

  KFrags kcur, knext;
  loadK(kcur, Kh, bh0, 0);
  for (int j = 0; j < SEQ / 32; ++j) {
#pragma unroll
    for (int hs = 0; hs < 2; ++hs) {
      // Preload the next (j,head) step's K-frags while this step computes.
      if (hs == 0)
        loadK(knext, Kh, bh0 + 1, j * 32);
      else if (j + 1 < SEQ / 32)
        loadK(knext, Kh, bh0, (j + 1) * 32);
      v8f s0 = zero8(), s1 = zero8();
      s0 = wmma_bf16(qa[hs][0], kcur.f[0], s0);
      s0 = wmma_bf16(qa[hs][1], kcur.f[1], s0);
      s1 = wmma_bf16(qa[hs][0], kcur.f[2], s1);
      s1 = wmma_bf16(qa[hs][1], kcur.f[3], s1);
      kcur = knext;
#pragma unroll
      for (int i = 0; i < 8; ++i) {
        const float a0 = s0[i] * scale, a1 = s1[i] * scale;
        float mx = fmaxf(a0, a1);
#pragma unroll
        for (int o = 8; o >= 1; o >>= 1) mx = fmaxf(mx, __shfl_xor(mx, o, 32));
        const float mn = fmaxf(m[hs][i], mx);
        float e = __expf(a0 - mn) + __expf(a1 - mn);
#pragma unroll
        for (int o = 8; o >= 1; o >>= 1) e += __shfl_xor(e, o, 32);
        l[hs][i] = l[hs][i] * __expf(m[hs][i] - mn) + e;
        m[hs][i] = mn;
      }
    }
  }

  // Fused softmax offset: exp(s - m)/l == exp(s - (m + ln l))
  float ofs[2][8];
#pragma unroll
  for (int hs = 0; hs < 2; ++hs)
#pragma unroll
    for (int i = 0; i < 8; ++i) ofs[hs][i] = m[hs][i] + __logf(l[hs][i]);

  // ---- Pass 2: probs -> LDS -> (avg over heads) + P@V accumulation
  v8f acc[2][4];
#pragma unroll
  for (int hs = 0; hs < 2; ++hs)
#pragma unroll
    for (int nt = 0; nt < 4; ++nt) acc[hs][nt] = zero8();

  loadK(kcur, Kh, bh0, 0);
  for (int j = 0; j < SEQ / 32; ++j) {
    const int kv0 = j * 32;

    // Hoist this iteration's V-frags: in flight across the softmax stores.
    v16bf vb[2][4];
#pragma unroll
    for (int hs = 0; hs < 2; ++hs)
#pragma unroll
      for (int nt = 0; nt < 4; ++nt)
        vb[hs][nt] = fragB_g(
            Vt + (((size_t)(bh0 + hs)) * DK + nt * 16) * SEQ + kv0, SEQ);

#pragma unroll
    for (int hs = 0; hs < 2; ++hs) {
      if (hs == 0)
        loadK(knext, Kh, bh0 + 1, kv0);
      else if (j + 1 < SEQ / 32)
        loadK(knext, Kh, bh0, kv0 + 32);
      v8f s0 = zero8(), s1 = zero8();
      s0 = wmma_bf16(qa[hs][0], kcur.f[0], s0);
      s0 = wmma_bf16(qa[hs][1], kcur.f[1], s0);
      s1 = wmma_bf16(qa[hs][0], kcur.f[2], s1);
      s1 = wmma_bf16(qa[hs][1], kcur.f[3], s1);
      kcur = knext;
      float* buf = s_p[wave * 2 + hs];
#pragma unroll
      for (int i = 0; i < 8; ++i) {
        const int row = i + half * 8;
        buf[row * 32 + ln]      = __expf(s0[i] * scale - ofs[hs][i]);
        buf[row * 32 + 16 + ln] = __expf(s1[i] * scale - ofs[hs][i]);
      }
    }
    __syncthreads();

#pragma unroll
    for (int hs = 0; hs < 2; ++hs) {
      v16bf pa = fragA_lds_f32(s_p[wave * 2 + hs]);
#pragma unroll
      for (int nt = 0; nt < 4; ++nt)
        acc[hs][nt] = wmma_bf16(pa, vb[hs][nt], acc[hs][nt]);
    }

    // avg_attention: mean over heads = mean over the 16 LDS buffers
    for (int e = threadIdx.x; e < 512; e += 256) {
      float ssum = 0.f;
#pragma unroll
      for (int k = 0; k < 16; ++k) ssum += s_p[k][e];
      const int row = e >> 5, col = e & 31;
      Avg[(((size_t)b) * SEQ + q0 + row) * SEQ + kv0 + col] =
          ssum * (1.0f / NHEADS);
    }
    __syncthreads();
  }

  // attended -> bf16 row-major [n, DMODEL], heads re-merged
#pragma unroll
  for (int hs = 0; hs < 2; ++hs) {
    const int h = wave * 2 + hs;
#pragma unroll
    for (int nt = 0; nt < 4; ++nt) {
#pragma unroll
      for (int i = 0; i < 8; ++i) {
        const int row = q0 + i + half * 8;
        const int col = h * DK + nt * 16 + ln;
        Att[((size_t)b * SEQ + row) * DMODEL + col] = (bf16)acc[hs][nt][i];
      }
    }
  }
}

// ---------- Host launcher ----------

extern "C" void kernel_launch(void* const* d_in, const int* in_sizes, int n_in,
                              void* d_out, int out_size, void* d_ws, size_t ws_size,
                              hipStream_t stream) {
  (void)in_sizes; (void)n_in; (void)out_size; (void)ws_size;

  const float* q  = (const float*)d_in[0];
  const float* k  = (const float*)d_in[1];
  const float* v  = (const float*)d_in[2];
  const float* wq = (const float*)d_in[3];
  const float* wk = (const float*)d_in[4];
  const float* wv = (const float*)d_in[5];
  const float* wo = (const float*)d_in[6];

  char* ws = (char*)d_ws;
  size_t off = 0;
  auto alloc = [&](size_t bytes) -> void* {
    void* p = ws + off;
    off += (bytes + 255) & ~(size_t)255;
    return p;
  };

  const size_t nX = (size_t)NROWS * DMODEL;     // 4 Mi elems
  const size_t nW = (size_t)DMODEL * DMODEL;    // 1 Mi elems

  bf16* Xq = (bf16*)alloc(nX * 2);
  bf16* Xk = (bf16*)alloc(nX * 2);
  bf16* Xv = (bf16*)alloc(nX * 2);
  bf16* Wq = (bf16*)alloc(nW * 2);
  bf16* Wk = (bf16*)alloc(nW * 2);
  bf16* Wv = (bf16*)alloc(nW * 2);
  bf16* Wo = (bf16*)alloc(nW * 2);
  bf16* Qh = (bf16*)alloc(nX * 2);   // [b,h,s,d]
  bf16* Kh = (bf16*)alloc(nX * 2);   // [b,h,s,d]
  bf16* Vt = (bf16*)alloc(nX * 2);   // [b,h,d,s]
  bf16* At = (bf16*)alloc(nX * 2);   // attended [n, DMODEL]

  const int TB = 256;
  cvt_kernel<<<(int)((nX + TB - 1) / TB), TB, 0, stream>>>(q,  Xq, (int)nX);
  cvt_kernel<<<(int)((nX + TB - 1) / TB), TB, 0, stream>>>(k,  Xk, (int)nX);
  cvt_kernel<<<(int)((nX + TB - 1) / TB), TB, 0, stream>>>(v,  Xv, (int)nX);
  cvt_kernel<<<(int)((nW + TB - 1) / TB), TB, 0, stream>>>(wq, Wq, (int)nW);
  cvt_kernel<<<(int)((nW + TB - 1) / TB), TB, 0, stream>>>(wk, Wk, (int)nW);
  cvt_kernel<<<(int)((nW + TB - 1) / TB), TB, 0, stream>>>(wv, Wv, (int)nW);
  cvt_kernel<<<(int)((nW + TB - 1) / TB), TB, 0, stream>>>(wo, Wo, (int)nW);

  dim3 gg(NROWS / 32, DMODEL / 512);
  gemm_kernel<<<gg, 256, 0, stream>>>(Xq, Wq, Qh, 0);
  gemm_kernel<<<gg, 256, 0, stream>>>(Xk, Wk, Kh, 0);
  gemm_kernel<<<gg, 256, 0, stream>>>(Xv, Wv, Vt, 1);

  float* avg_out = (float*)d_out + nX;  // output first, then avg_attention
  dim3 ga(SEQ / 16, BATCH);
  attn_kernel<<<ga, 256, 0, stream>>>(Qh, Kh, Vt, At, avg_out);

  gemm_kernel<<<gg, 256, 0, stream>>>(At, Wo, d_out, 2);
}
